// PointSetTracker_15066745275076
// MI455X (gfx1250) — compile-verified, matrix operand order checked by reading, not run on hardware
//
#include <hip/hip_runtime.h>
#include <hip/hip_bf16.h>

// ---------------- problem constants ----------------
#define BATCH 64
#define KPTS  512
#define SD    128
#define NL    8
#define MR    (BATCH * KPTS)          // 32768 rows in (B*K, ch) layout
#define STCH  1536                    // [h_new(256) | g(256) | states 8*128]

typedef __attribute__((ext_vector_type(16))) __bf16 v16bf;
typedef __attribute__((ext_vector_type(8)))  float  v8f;

// ---------------- TDM availability ----------------
#if defined(__has_builtin)
#if __has_builtin(__builtin_amdgcn_tensor_load_to_lds) && __has_builtin(__builtin_amdgcn_s_wait_tensorcnt)
#define HAVE_TDM 1
#endif
#endif

#if defined(HAVE_TDM)
#warning "CDNA5: tensor_load_to_lds (TDM) path ENABLED"
typedef __attribute__((ext_vector_type(4))) unsigned int u32x4;
typedef __attribute__((ext_vector_type(8))) int          i32x8;
typedef __attribute__((ext_vector_type(4))) int          i32x4;

// DMA `rows` rows of 1024 dwords (4KB each), row stride strideDw dwords,
// from global into contiguous LDS via the Tensor Data Mover; drain TENSORcnt.
__device__ __forceinline__ void tdm_copy(const unsigned short* src, unsigned short* dstLds,
                                         int rows, int strideDw)
{
    unsigned long long ga = (unsigned long long)(uintptr_t)src;
    unsigned int       la = (unsigned int)(uintptr_t)dstLds;
    u32x4 g0;
    g0.x = 1u;                                                    // count=1, is_restore=0
    g0.y = la;                                                    // lds_addr
    g0.z = (unsigned int)ga;                                      // global_addr[31:0]
    g0.w = (unsigned int)((ga >> 32) & 0x1FFFFFFull) | (2u << 30);// addr[56:32] | type=2
    i32x8 g1;
    g1[0] = 0x20000;                   // workgroup_mask=0, data_size=2 (4 bytes)
    g1[1] = (int)(1024u << 16);        // tensor_dim0 lo16 = 1024 (bits 63:48)
    g1[2] = (int)((unsigned)rows << 16); // tensor_dim0 hi=0 | tensor_dim1 lo16 = rows
    g1[3] = (int)(1024u << 16);        // tensor_dim1 hi=0 | tile_dim0 = 1024
    g1[4] = rows;                      // tile_dim1 = rows, tile_dim2 = 0
    g1[5] = strideDw;                  // tensor_dim0_stride lo32
    g1[6] = 0;
    g1[7] = 0;
    i32x4 z4 = {0, 0, 0, 0};
#if __clang_major__ >= 23
    i32x8 z8 = {0, 0, 0, 0, 0, 0, 0, 0};
    __builtin_amdgcn_tensor_load_to_lds(g0, g1, z4, z4, z8, 0);
#else
    __builtin_amdgcn_tensor_load_to_lds(g0, g1, z4, z4, 0);
#endif
    __builtin_amdgcn_s_wait_tensorcnt(0);
}
#else
#warning "CDNA5: TDM builtins not found - using LDS uint4 copy fallback"
#endif

// ---------------- helpers ----------------
__device__ __forceinline__ unsigned short f2bf(float f) {
    union { float f; unsigned int u; } v; v.f = f;
    unsigned int u = v.u;
    return (unsigned short)((u + 0x7FFFu + ((u >> 16) & 1u)) >> 16);
}
// packed f32x2 -> bf16x2 via the hip header (hardware packed cvt where available)
__device__ __forceinline__ unsigned int pack2bf(float a, float b) {
    union { __hip_bfloat162 h; unsigned int u; } cv;
    cv.h = __float22bfloat162_rn(make_float2(a, b));
    return cv.u;
}

// ISA 16-bit fragment layout for a 16x32 tile (A: r=row, B: r=col):
// lanes 0-15 hold k{0-7,16-23}, lanes 16-31 hold k{8-15,24-31}
__device__ __forceinline__ int fragIdx(int r, int k) {
    int lane = r + (((k >> 3) & 1) << 4);
    int e = (k & 7) | ((k & 16) >> 1);
    return lane * 16 + e;
}

__device__ __forceinline__ float sigmf(float x) { return 1.0f / (1.0f + __expf(-x)); }

// ---------------- weight pre-pack: f32 -> fragment-swizzled bf16 4KB tiles ----------------
// tile order: tile = nt*(Kd/32)+kt, element (kk,n) at ((n>>4)<<9)+fragIdx(n&15,kk)
// mode 0: W stored (Kd,N) row-major; mode 1: (N,Kd); mode 2: conv cw (128,128,9)
__global__ __launch_bounds__(256) void pack_w(
    const float* __restrict__ W, unsigned short* __restrict__ out,
    int Kd, int N, int ldw, int mode)
{
    long i = (long)blockIdx.x * 256 + threadIdx.x;
    if (i >= (long)Kd * N) return;
    long tile = i >> 11; int pos = (int)(i & 2047);
    int ktiles = Kd >> 5;
    int kt = (int)(tile % ktiles), nt = (int)(tile / ktiles);
    int kk = pos >> 6, n = pos & 63;
    int kg = kt * 32 + kk, ng = nt * 64 + n;
    float v;
    if (mode == 0)      v = W[(long)kg * ldw + ng];
    else if (mode == 1) v = W[(long)ng * ldw + kg];
    else                v = W[((long)ng * 128 + (kg & 127)) * 9 + (kg >> 7)];
    out[tile * 2048 + ((n >> 4) << 9) + fragIdx(n & 15, kk)] = f2bf(v);
}

// ---------------- generic bf16 WMMA GEMM, block tile 128 x NT ----------------
// Y[M,N] = act(alpha * X[M,Kd] @ W + bias) (+Y if ACCUM)
// PACKEDW: pre-swizzled weight tiles, staged by TDM (or raw uint4 copies).
// else TRANSW selects (N,Kd) vs (Kd,N) f32 with on-the-fly bf16 conversion.
template <int NT, bool TRANSW, bool RELU, bool ACCUM, bool BIAS, bool PACKEDW>
__global__ __launch_bounds__(256) void gemm_wmma(
    const float* __restrict__ X, const float* __restrict__ W,
    const unsigned short* __restrict__ Wpk,
    const float* __restrict__ bias, float* __restrict__ Y,
    int M, int N, int Kd, int ldx, int ldw, int ldy,
    long sXb, long sXh, long sWb, long sWh, long sYb, long sYh,
    int nH, float alpha)
{
    constexpr int NF = NT / 32;                    // b-fragments per wave
    __shared__ __align__(16) unsigned short As[8 * 32 * 16];       // 128x32 bf16
    __shared__ __align__(16) unsigned short Bs[(NT / 16) * 32 * 16]; // 32xNT bf16

    int z = blockIdx.z, bb = z / nH, hh = z % nH;
    X += (long)bb * sXb + (long)hh * sXh;
    if (!PACKEDW) W += (long)bb * sWb + (long)hh * sWh;
    Y += (long)bb * sYb + (long)hh * sYh;

    int n0 = blockIdx.x * NT, m0 = blockIdx.y * 128;
    int tid = threadIdx.x, lane = tid & 31, w = tid >> 5;
    int wm = w & 3, wn = w >> 2;                   // 4x2 wave grid -> 32 x NT/2 per wave

    v8f acc[2][NF] = {};

    for (int k0 = 0; k0 < Kd; k0 += 32) {
        if (k0 + 32 < Kd)
            __builtin_prefetch(&X[(long)(m0 + (tid & 127)) * ldx + k0 + 32], 0, 1);
        // stage A: 128 rows x 32 k as 2048 bf16 pairs (8 per thread, b32 stores)
        #pragma unroll
        for (int i = 0; i < 8; ++i) {
            int pidx = tid * 8 + i; int r = pidx >> 4, k = (pidx & 15) * 2;
            const float* p = &X[(long)(m0 + r) * ldx + k0 + k];
            *(unsigned int*)&As[((r >> 4) << 9) + fragIdx(r & 15, k)] = pack2bf(p[0], p[1]);
        }
        // stage B
        if (PACKEDW) {
            int ktiles = Kd >> 5;
            const unsigned short* tile =
                Wpk + (((size_t)(n0 >> 6) * (size_t)ktiles) + (size_t)(k0 >> 5)) * 2048;
#if defined(HAVE_TDM)
            if (w == 0) tdm_copy(tile, Bs, NT / 64, ktiles * 1024);
#else
            #pragma unroll
            for (int r = 0; r < NT / 64; ++r)
                ((uint4*)Bs)[r * 256 + tid] =
                    ((const uint4*)(tile + (size_t)r * ktiles * 2048))[tid];
#endif
        } else {
            #pragma unroll
            for (int i = 0; i < NT / 16; ++i) {
                int pidx = tid * (NT / 16) + i; int n = pidx >> 4, kk = (pidx & 15) * 2;
                float w0, w1;
                if (TRANSW) {
                    const float* p = &W[(long)(n0 + n) * ldw + k0 + kk];
                    w0 = p[0]; w1 = p[1];
                } else {
                    w0 = W[(long)(k0 + kk) * ldw + n0 + n];
                    w1 = W[(long)(k0 + kk + 1) * ldw + n0 + n];
                }
                *(unsigned int*)&Bs[((n >> 4) << 9) + fragIdx(n & 15, kk)] = pack2bf(w0, w1);
            }
        }
        __syncthreads();
        v16bf a0 = *(const v16bf*)&As[((wm * 2 + 0) << 9) + lane * 16];
        v16bf a1 = *(const v16bf*)&As[((wm * 2 + 1) << 9) + lane * 16];
        #pragma unroll
        for (int t = 0; t < NF; ++t) {
            v16bf b = *(const v16bf*)&Bs[((wn * NF + t) << 9) + lane * 16];
            acc[0][t] = __builtin_amdgcn_wmma_f32_16x16x32_bf16(false, a0, false, b, (short)0, acc[0][t], false, false);
            acc[1][t] = __builtin_amdgcn_wmma_f32_16x16x32_bf16(false, a1, false, b, (short)0, acc[1][t], false, false);
        }
        __syncthreads();
    }

    int rl = (lane & 16) >> 1;   // +8 for upper half-wave
    int cl = lane & 15;
    #pragma unroll
    for (int s = 0; s < 2; ++s)
    #pragma unroll
    for (int t = 0; t < NF; ++t)
    #pragma unroll
    for (int j = 0; j < 8; ++j) {
        int r = m0 + wm * 32 + s * 16 + j + rl;
        int c = n0 + (wn * NF + t) * 16 + cl;
        float v = acc[s][t][j] * alpha;
        if (BIAS) v += bias[c];
        if (RELU) v = fmaxf(v, 0.0f);
        long off = (long)r * ldy + c;
        if (ACCUM) v += Y[off];
        Y[off] = v;
    }
}

// ---------------- circular dilated conv as im2col-on-the-fly WMMA GEMM ----------------
// out[m, o] = relu( sum_{t,c} cw[o,c,t] * Yin[b, (k+(t-4)*dil) mod 512, c] + cb[o] ) (+res)
__global__ __launch_bounds__(256) void conv_gemm(
    const float* __restrict__ Yin,           // (B*K, 128)
    const unsigned short* __restrict__ Pcw,  // packed conv weights, Kd=1152, N=128
    const float* __restrict__ cb,
    const float* __restrict__ res, int ldres,
    float* __restrict__ out, int ldo, int dil)
{
    __shared__ __align__(16) unsigned short As[8 * 32 * 16];
    __shared__ __align__(16) unsigned short Bs[8 * 32 * 16];   // 32x128 bf16

    int m0 = blockIdx.y * 128;                 // N covered fully (128)
    int tid = threadIdx.x, lane = tid & 31, w = tid >> 5;
    int wm = w & 3, wn = w >> 2;
    v8f acc[2][4] = {};

    for (int k0 = 0; k0 < 1152; k0 += 32) {       // Kd = 9 taps * 128 ch
        int tap = k0 >> 7;                        // 32-chunk never straddles a tap
        int c0 = k0 & 127;
        int shift = (tap - 4) * dil;
        #pragma unroll
        for (int i = 0; i < 8; ++i) {
            int pidx = tid * 8 + i; int r = pidx >> 4, k = (pidx & 15) * 2;
            int m = m0 + r, b = m >> 9, kp = m & 511;
            int ks = (kp + shift + 512) & 511;    // circular in K
            const float* p = &Yin[((long)((b << 9) | ks)) * 128 + c0 + k];
            *(unsigned int*)&As[((r >> 4) << 9) + fragIdx(r & 15, k)] = pack2bf(p[0], p[1]);
        }
        const unsigned short* tile = Pcw + (size_t)(k0 >> 5) * 2048;
#if defined(HAVE_TDM)
        if (w == 0) tdm_copy(tile, Bs, 2, 36 * 1024);
#else
        #pragma unroll
        for (int r = 0; r < 2; ++r)
            ((uint4*)Bs)[r * 256 + tid] =
                ((const uint4*)(tile + (size_t)r * 36 * 2048))[tid];
#endif
        __syncthreads();
        v16bf a0 = *(const v16bf*)&As[((wm * 2 + 0) << 9) + lane * 16];
        v16bf a1 = *(const v16bf*)&As[((wm * 2 + 1) << 9) + lane * 16];
        #pragma unroll
        for (int t = 0; t < 4; ++t) {
            v16bf b = *(const v16bf*)&Bs[((wn * 4 + t) << 9) + lane * 16];
            acc[0][t] = __builtin_amdgcn_wmma_f32_16x16x32_bf16(false, a0, false, b, (short)0, acc[0][t], false, false);
            acc[1][t] = __builtin_amdgcn_wmma_f32_16x16x32_bf16(false, a1, false, b, (short)0, acc[1][t], false, false);
        }
        __syncthreads();
    }

    int rl = (lane & 16) >> 1, cl = lane & 15;
    #pragma unroll
    for (int s = 0; s < 2; ++s)
    #pragma unroll
    for (int t = 0; t < 4; ++t)
    #pragma unroll
    for (int j = 0; j < 8; ++j) {
        int r = m0 + wm * 32 + s * 16 + j + rl;
        int c = (wn * 4 + t) * 16 + cl;
        float v = fmaxf(acc[s][t][j] + cb[c], 0.0f);   // relu THEN residual
        if (res) v += res[(long)r * ldres + c];
        out[(long)r * ldo + c] = v;
    }
}

// ---------------- layernorm over 128 channels, one wave per row ----------------
__global__ __launch_bounds__(256) void ln_kernel(
    const float* __restrict__ X, int ldx,
    const float* __restrict__ g, const float* __restrict__ b,
    float* __restrict__ out, int ldo)
{
    int row = blockIdx.x * 8 + (threadIdx.x >> 5);
    int lane = threadIdx.x & 31;
    const float* xr = X + (long)row * ldx;
    float v[4], s = 0.0f;
    #pragma unroll
    for (int i = 0; i < 4; ++i) { v[i] = xr[lane * 4 + i]; s += v[i]; }
    #pragma unroll
    for (int m = 16; m >= 1; m >>= 1) s += __shfl_xor(s, m, 32);
    float mu = s * (1.0f / 128.0f), q = 0.0f;
    #pragma unroll
    for (int i = 0; i < 4; ++i) { float d = v[i] - mu; q += d * d; }
    #pragma unroll
    for (int m = 16; m >= 1; m >>= 1) q += __shfl_xor(q, m, 32);
    float rs = rsqrtf(q * (1.0f / 128.0f) + 1e-6f);
    float* o = out + (long)row * ldo;
    #pragma unroll
    for (int i = 0; i < 4; ++i) {
        int c = lane * 4 + i;
        o[c] = (v[i] - mu) * rs * g[c] + b[c];
    }
}

// ---------------- softmax over rows of length 512, one wave per row ----------------
__global__ __launch_bounds__(256) void softmax512(float* __restrict__ S)
{
    long row = (long)blockIdx.x * 8 + (threadIdx.x >> 5);
    int lane = threadIdx.x & 31;
    float* r = S + row * 512;
    float v[16], mx = -1e30f;
    #pragma unroll
    for (int j = 0; j < 16; ++j) { v[j] = r[lane + 32 * j]; mx = fmaxf(mx, v[j]); }
    #pragma unroll
    for (int m = 16; m >= 1; m >>= 1) mx = fmaxf(mx, __shfl_xor(mx, m, 32));
    float s = 0.0f;
    #pragma unroll
    for (int j = 0; j < 16; ++j) { v[j] = __expf(v[j] - mx); s += v[j]; }
    #pragma unroll
    for (int m = 16; m >= 1; m >>= 1) s += __shfl_xor(s, m, 32);
    float inv = 1.0f / s;
    #pragma unroll
    for (int j = 0; j < 16; ++j) r[lane + 32 * j] = v[j] * inv;
}

// ---------------- (B,C,K) -> (B*K, C) transpose ----------------
__global__ __launch_bounds__(256) void transp_bck(
    const float* __restrict__ x, float* __restrict__ xt)
{
    long i = (long)blockIdx.x * 256 + threadIdx.x;
    if (i >= (long)BATCH * SD * KPTS) return;
    int k = (int)(i & 511), c = (int)((i >> 9) & 127), b = (int)(i >> 16);
    xt[((long)((b << 9) | k)) * 128 + c] = x[i];
}

// ---------------- fusion: g = max_k(tmp), broadcast into state cols 256..511 ----------------
__global__ __launch_bounds__(256) void fusmax(
    const float* __restrict__ tmp,   // (B*K, 256)
    float* __restrict__ stateAll)    // (B*K, 1536)
{
    int b = blockIdx.x, ch = threadIdx.x;
    float m = -1e30f;
    for (int k = 0; k < 512; ++k)
        m = fmaxf(m, tmp[((long)((b << 9) | k)) * 256 + ch]);
    for (int k = 0; k < 512; ++k)
        stateAll[((long)((b << 9) | k)) * STCH + 256 + ch] = m;
}

// ---------------- LSTM cell elementwise ----------------
__global__ __launch_bounds__(256) void lstm_kernel(
    const float* __restrict__ gates,   // (B*K, 1024) = [i|f|g|o]
    const float* __restrict__ c0,      // (B*K, 256)
    float* __restrict__ hout, float* __restrict__ cout,
    float* __restrict__ stateAll)
{
    long m = (long)blockIdx.x * 256 + threadIdx.x;
    long row = m >> 8; int c = (int)(m & 255);
    const float* gr = gates + row * 1024;
    float gi = sigmf(gr[c]);
    float gf = sigmf(gr[256 + c]);
    float gg = tanhf(gr[512 + c]);
    float go = sigmf(gr[768 + c]);
    float cn = gf * c0[m] + gi * gg;
    float hn = go * tanhf(cn);
    hout[m] = hn; cout[m] = cn;
    stateAll[row * STCH + c] = hn;
}

// ---------------- final N=2 head with transposed store: (B*K,64) -> (B,2,K) ----------------
__global__ __launch_bounds__(256) void p3_kernel(
    const float* __restrict__ p2o, const float* __restrict__ w,
    const float* __restrict__ bias, float* __restrict__ out)
{
    long m = (long)blockIdx.x * 256 + threadIdx.x;
    if (m >= (long)MR) return;
    int b = (int)(m >> 9), k = (int)(m & 511);
    const float* xr = p2o + m * 64;
    #pragma unroll
    for (int j = 0; j < 2; ++j) {
        float s = bias[j];
        for (int c = 0; c < 64; ++c) s += w[j * 64 + c] * xr[c];
        out[((long)b * 2 + j) * 512 + k] = s;
    }
}

// ---------------- host-side launchers ----------------
template <int NT, bool TW, bool RL, bool AC, bool BI, bool PK>
static void launch_gemm(hipStream_t st, const float* X, const float* W,
                        const unsigned short* Wpk, const float* bias, float* Y,
                        int M, int N, int Kd, int ldx, int ldw, int ldy,
                        long sXb = 0, long sXh = 0, long sWb = 0, long sWh = 0,
                        long sYb = 0, long sYh = 0, int nH = 1, int nZ = 1,
                        float alpha = 1.0f)
{
    dim3 grid(N / NT, M / 128, nZ);
    gemm_wmma<NT, TW, RL, AC, BI, PK><<<grid, 256, 0, st>>>(
        X, W, Wpk, bias, Y, M, N, Kd, ldx, ldw, ldy,
        sXb, sXh, sWb, sWh, sYb, sYh, nH, alpha);
}

static void launch_pack(hipStream_t st, const float* W, unsigned short* out,
                        int Kd, int N, int ldw, int mode)
{
    long n = (long)Kd * N;
    pack_w<<<(unsigned)((n + 255) / 256), 256, 0, st>>>(W, out, Kd, N, ldw, mode);
}

extern "C" void kernel_launch(void* const* d_in, const int* in_sizes, int n_in,
                              void* d_out, int out_size, void* d_ws, size_t ws_size,
                              hipStream_t stream)
{
    const float* x    = (const float*)d_in[0];
    const float* h0   = (const float*)d_in[1];
    const float* c0   = (const float*)d_in[2];
    const float* ln_g = (const float*)d_in[3];
    const float* ln_b = (const float*)d_in[4];
    const float* Wq   = (const float*)d_in[5];
    const float* Wk   = (const float*)d_in[6];
    const float* Wv   = (const float*)d_in[7];
    const float* Ww   = (const float*)d_in[8];
    const float* Wa   = (const float*)d_in[9];
    const float* cw   = (const float*)d_in[10];
    const float* cb   = (const float*)d_in[11];
    const float* fus_w = (const float*)d_in[12];
    const float* fus_b = (const float*)d_in[13];
    const float* rs_w  = (const float*)d_in[14];
    const float* rs_b  = (const float*)d_in[15];
    const float* W_ih  = (const float*)d_in[16];
    const float* W_hh  = (const float*)d_in[17];
    const float* b_ih  = (const float*)d_in[18];
    const float* b_hh  = (const float*)d_in[19];
    const float* p1_w  = (const float*)d_in[20];
    const float* p1_b  = (const float*)d_in[21];
    const float* p2_w  = (const float*)d_in[22];
    const float* p2_b  = (const float*)d_in[23];
    const float* p3_w  = (const float*)d_in[24];
    const float* p3_b  = (const float*)d_in[25];

    float* ws = (float*)d_ws;
    size_t off = 0;
    auto alloc = [&](size_t n) { float* p = ws + off; off += n; return p; };
    float* XT = alloc((size_t)MR * 128);
    float* LN = alloc((size_t)MR * 128);     // also reused for r@Ww result
    float* QB = alloc((size_t)MR * 128);
    float* KB = alloc((size_t)MR * 128);
    float* VB = alloc((size_t)MR * 512);
    float* SC = alloc((size_t)256 * 512 * 512);
    float* RB = alloc((size_t)MR * 512);
    float* YB = alloc((size_t)MR * 128);
    float* ST = alloc((size_t)MR * STCH);
    float* TF = alloc((size_t)MR * 256);
    float* RN = alloc((size_t)MR * 256);
    float* GT = alloc((size_t)MR * 1024);
    float* P1 = alloc((size_t)MR * 256);
    float* P2 = alloc((size_t)MR * 64);

    // packed-weight arena (bf16, fragment-swizzled tiles) after float scratch
    unsigned short* wsu = (unsigned short*)(ws + off);
    size_t uoff = 0;
    auto ualloc = [&](size_t n) { unsigned short* p = wsu + uoff; uoff += n; return p; };
    unsigned short* PWq[NL]; unsigned short* PWk[NL]; unsigned short* PWv[NL];
    unsigned short* PWw[NL]; unsigned short* PWa0[NL]; unsigned short* PWa1[NL];
    unsigned short* Pcw[NL];
    for (int i = 0; i < NL; ++i) {
        PWq[i]  = ualloc(128 * 128);
        PWk[i]  = ualloc(128 * 128);
        PWv[i]  = ualloc(128 * 512);
        PWw[i]  = ualloc(512 * 128);
        PWa0[i] = ualloc(128 * 128);
        PWa1[i] = ualloc(128 * 128);
        Pcw[i]  = ualloc(1152 * 128);
    }
    unsigned short* Pfus = ualloc(1024 * 256);
    unsigned short* Prs  = ualloc(1280 * 256);
    unsigned short* Pih  = ualloc(256 * 1024);
    unsigned short* Phh  = ualloc(256 * 1024);
    unsigned short* Pp1  = ualloc(1536 * 256);
    unsigned short* Pp2  = ualloc(256 * 64);
    (void)ws_size; (void)in_sizes; (void)n_in; (void)out_size;

    const int DIL[NL] = {1, 1, 1, 1, 2, 2, 4, 4};
    const float inv_sqrt_d = 0.17677669529663688f;   // 1/sqrt(32)

    // -------- one-time per-launch weight packing --------
    for (int i = 0; i < NL; ++i) {
        launch_pack(stream, Wq + (size_t)i * 128 * 128, PWq[i], 128, 128, 128, 0);
        launch_pack(stream, Wk + (size_t)i * 128 * 128, PWk[i], 128, 128, 128, 0);
        launch_pack(stream, Wv + (size_t)i * 128 * 512, PWv[i], 128, 512, 512, 0);
        launch_pack(stream, Ww + (size_t)i * 512 * 128, PWw[i], 512, 128, 128, 0);
        launch_pack(stream, Wa + (size_t)i * 256 * 128, PWa0[i], 128, 128, 128, 0);
        launch_pack(stream, Wa + (size_t)i * 256 * 128 + 128 * 128, PWa1[i], 128, 128, 128, 0);
        launch_pack(stream, cw + (size_t)i * 128 * 128 * 9, Pcw[i], 1152, 128, 0, 2);
    }
    launch_pack(stream, fus_w, Pfus, 1024, 256, 1024, 1);
    launch_pack(stream, rs_w,  Prs,  1280, 256, 1280, 1);
    launch_pack(stream, W_ih,  Pih,  256, 1024, 256, 1);
    launch_pack(stream, W_hh,  Phh,  256, 1024, 256, 1);
    launch_pack(stream, p1_w,  Pp1,  1536, 256, 1536, 1);
    launch_pack(stream, p2_w,  Pp2,  256, 64, 256, 1);

    // (B,C,K) -> (B*K, C)
    {
        long n = (long)BATCH * SD * KPTS;
        transp_bck<<<(unsigned)((n + 255) / 256), 256, 0, stream>>>(x, XT);
    }

    for (int i = 0; i < NL; ++i) {
        const float* hin = (i == 0) ? XT : (ST + 512 + 128 * (i - 1));
        int ldin = (i == 0) ? 128 : STCH;

        ln_kernel<<<MR / 8, 256, 0, stream>>>(hin, ldin, ln_g + i * 128, ln_b + i * 128, LN, 128);

        // q, k, v projections (packed weights, TDM-staged)
        launch_gemm<128, false, false, false, false, true>(stream, LN, nullptr, PWq[i], nullptr, QB,
                                                           MR, 128, 128, 128, 0, 128);
        launch_gemm<128, false, false, false, false, true>(stream, LN, nullptr, PWk[i], nullptr, KB,
                                                           MR, 128, 128, 128, 0, 128);
        launch_gemm<128, false, false, false, false, true>(stream, LN, nullptr, PWv[i], nullptr, VB,
                                                           MR, 512, 128, 128, 0, 512);

        // scores[b,h] = q @ k^T * 1/sqrt(d), batched over z = b*4+h (activation weights: unpacked)
        launch_gemm<128, true, false, false, false, false>(stream, QB, KB, nullptr, nullptr, SC,
                                                           512, 512, 32, 128, 128, 512,
                                                           (long)512 * 128, 32,
                                                           (long)512 * 128, 32,
                                                           (long)4 * 512 * 512, (long)512 * 512,
                                                           4, 256, inv_sqrt_d);

        softmax512<<<(unsigned)((256L * 512) / 8), 256, 0, stream>>>(SC);

        // r[b,h] = attn @ v  -> RB (B*K, 512) head-concat
        launch_gemm<128, false, false, false, false, false>(stream, SC, VB, nullptr, nullptr, RB,
                                                            512, 128, 512, 512, 512, 512,
                                                            (long)4 * 512 * 512, (long)512 * 512,
                                                            (long)512 * 512, 128,
                                                            (long)512 * 512, 128,
                                                            4, 256);

        // rw = r @ Ww  (512 -> 128), into LN (free now)
        launch_gemm<128, false, false, false, false, true>(stream, RB, nullptr, PWw[i], nullptr, LN,
                                                           MR, 128, 512, 512, 0, 128);

        // y = [rw, res] @ Wa as two accumulating GEMMs
        launch_gemm<128, false, false, false, false, true>(stream, LN, nullptr, PWa0[i], nullptr, YB,
                                                           MR, 128, 128, 128, 0, 128);
        launch_gemm<128, false, false, true, false, true>(stream, hin, nullptr, PWa1[i], nullptr, YB,
                                                          MR, 128, 128, ldin, 0, 128);

        // dilated circular conv + relu + residual -> state slot i
        dim3 cgrid(1, MR / 128);
        conv_gemm<<<cgrid, 256, 0, stream>>>(
            YB, Pcw[i], cb + (size_t)i * 128,
            (i == 0) ? nullptr : hin, ldin,
            ST + 512 + 128 * i, STCH, DIL[i]);
    }

    // fusion: tmp = state(1024) @ fus_w^T + b ; g = max_k ; broadcast into cols 256..511
    launch_gemm<128, false, false, false, true, true>(stream, ST + 512, nullptr, Pfus, fus_b, TF,
                                                      MR, 256, 1024, STCH, 0, 256);
    fusmax<<<BATCH, 256, 0, stream>>>(TF, ST);

    // rnn_in = [g|states](1280) @ rs_w^T + b
    launch_gemm<128, false, false, false, true, true>(stream, ST + 256, nullptr, Prs, rs_b, RN,
                                                      MR, 256, 1280, STCH, 0, 256);

    // gates = rnn_in @ W_ih^T + b_ih + h0 @ W_hh^T + b_hh
    launch_gemm<128, false, false, false, true, true>(stream, RN, nullptr, Pih, b_ih, GT,
                                                      MR, 1024, 256, 256, 0, 1024);
    launch_gemm<128, false, false, true, true, true>(stream, h0, nullptr, Phh, b_hh, GT,
                                                     MR, 1024, 256, 256, 0, 1024);

    float* hout = (float*)d_out + (size_t)BATCH * 2 * KPTS;
    float* cout = hout + (size_t)MR * 256;
    lstm_kernel<<<(unsigned)(((size_t)MR * 256) / 256), 256, 0, stream>>>(GT, c0, hout, cout, ST);

    // prediction head
    launch_gemm<128, false, true, false, true, true>(stream, ST, nullptr, Pp1, p1_b, P1,
                                                     MR, 256, 1536, STCH, 0, 256);
    launch_gemm<64, false, true, false, true, true>(stream, P1, nullptr, Pp2, p2_b, P2,
                                                    MR, 64, 256, 256, 0, 64);
    p3_kernel<<<MR / 256, 256, 0, stream>>>(P2, p3_w, p3_b, (float*)d_out);
}
